// unit_gcn_requiredA_86852828659762
// MI455X (gfx1250) — compile-verified, matrix operand order checked by reading, not run on hardware
//
#include <hip/hip_runtime.h>

// ---------------------------------------------------------------------------
// unit-GCN block fused as one bf16 WMMA GEMM on gfx1250 (MI455X, wave32)
//   z[(n,t),(c,w)] = sum_{ci,v} x[(n,t),(ci,v)] * Weff[(c,w),(ci,v)] + bias_eff
//   Weff[(c,w),(ci,v)] = sum_k conv_w[k*64+c, ci] * A[k,v,w]
// then deterministic batch-stat BN (biased var) + ReLU.
// GEMM uses double-buffered LDS fed by GLOBAL_LOAD_ASYNC_TO_LDS (ASYNCcnt).
// ---------------------------------------------------------------------------

typedef __attribute__((ext_vector_type(16))) __bf16 v16bf;
typedef __attribute__((ext_vector_type(8)))  float  v8f;
typedef __attribute__((ext_vector_type(4)))  int    v4i;

union FragBF {
  v16bf v;
  uint4 u[2];
};

#define NB     64               // batch
#define CIN    64
#define COUT   64
#define TLEN   256
#define VJ     25
#define MDIM   (NB * TLEN)      // 16384 GEMM rows  (n,t)
#define KQDIM  (CIN * VJ)       // 1600  GEMM K     (ci,v)
#define NJDIM  (COUT * VJ)      // 1600  GEMM cols  (c,w)
#define CHCNT  (NB * TLEN * VJ) // 409600 elements per channel for BN
#define ELEMS  ((size_t)NB * COUT * TLEN * VJ) // 26,214,400 output f32

static __device__ __forceinline__ unsigned short f32_to_bf16_rne(float f) {
  unsigned int u = __float_as_uint(f);
  u += 0x7FFFu + ((u >> 16) & 1u);   // round-to-nearest-even
  return (unsigned short)(u >> 16);
}

// --- CDNA5 async global->LDS (16B per lane), ASYNCcnt tracked --------------
static __device__ __forceinline__ void async_copy_b128(unsigned short* lds_dst,
                                                       const unsigned short* gsrc) {
#if __has_builtin(__builtin_amdgcn_global_load_async_to_lds_b128)
  __builtin_amdgcn_global_load_async_to_lds_b128(
      (__attribute__((address_space(1))) v4i*)gsrc,
      (__attribute__((address_space(3))) v4i*)lds_dst, 0, 0);
#else
  unsigned int laddr = (unsigned int)(unsigned long long)
      (__attribute__((address_space(3))) unsigned short*)lds_dst;
  asm volatile("global_load_async_to_lds_b128 %0, %1, off"
               :: "v"(laddr), "v"(gsrc) : "memory");
#endif
}

static __device__ __forceinline__ void wait_async0() {
#if __has_builtin(__builtin_amdgcn_s_wait_asynccnt)
  __builtin_amdgcn_s_wait_asynccnt(0);
#else
  asm volatile("s_wait_asynccnt 0" ::: "memory");
#endif
}

// --- K0: x (n,ci,t,v) f32  ->  Xb[p][q] bf16,  p = n*256+t, q = ci*25+v ----
__global__ __launch_bounds__(256) void gcn_convert_x(const float* __restrict__ x,
                                                     unsigned short* __restrict__ Xb) {
  size_t idx = (size_t)blockIdx.x * 256 + threadIdx.x;
  if (idx >= (size_t)MDIM * KQDIM) return;
  int p  = (int)(idx / KQDIM);
  int q  = (int)(idx - (size_t)p * KQDIM);
  int n  = p >> 8, t = p & 255;
  int ci = q / VJ, v = q - ci * VJ;
  Xb[idx] = f32_to_bf16_rne(x[((size_t)(n * CIN + ci) * TLEN + t) * VJ + v]);
}

// --- K1: fused weights WeffT[j][q] bf16 (K-major) + f32 effective bias -----
__global__ __launch_bounds__(256) void gcn_build_weff(const float* __restrict__ conv_w,
                                                      const float* __restrict__ conv_b,
                                                      const float* __restrict__ A,
                                                      unsigned short* __restrict__ Wt,
                                                      float* __restrict__ bias) {
  int idx = blockIdx.x * 256 + threadIdx.x;
  if (idx >= NJDIM * KQDIM) return;
  int j  = idx / KQDIM;
  int q  = idx - j * KQDIM;
  int c  = j / VJ,  w = j - c * VJ;
  int ci = q / VJ,  v = q - ci * VJ;
  float s = 0.f;
#pragma unroll
  for (int k = 0; k < 3; ++k)
    s += conv_w[(k * COUT + c) * CIN + ci] * A[(k * VJ + v) * VJ + w];
  Wt[(size_t)j * KQDIM + q] = f32_to_bf16_rne(s);
  if (q == 0) {            // one thread per output column builds the bias
    float b = 0.f;
#pragma unroll
    for (int k = 0; k < 3; ++k) {
      float as = 0.f;
      for (int vv = 0; vv < VJ; ++vv) as += A[(k * VJ + vv) * VJ + w];
      b += conv_b[k * COUT + c] * as;
    }
    bias[j] = b;
  }
}

// --- K2: bf16 WMMA GEMM, block tile 256x64, 8 waves x (2x4) 16x16 tiles ----
// Double-buffered LDS, async global->LDS staging overlapped with WMMA.
#define BM 256
#define BN 64
#define PITCH 40   // 32 K elems + 8 pad -> 80B rows, conflict-free frag reads
__global__ __launch_bounds__(256) void gcn_gemm_wmma(const unsigned short* __restrict__ Xb,
                                                     const unsigned short* __restrict__ Wt,
                                                     const float* __restrict__ bias,
                                                     float* __restrict__ z) {
  __shared__ __align__(16) unsigned short As[2][BM * PITCH];
  __shared__ __align__(16) unsigned short Bs[2][BN * PITCH];

  const int tid  = threadIdx.x;
  const int lane = tid & 31;
  const int wid  = tid >> 5;
  const int pBase = blockIdx.x * BM;
  const int jBase = blockIdx.y * BN;
  const int wm = wid * 32;           // wave row offset; every wave spans all 64 cols

  // ISA 16-bit fragment layout: lanes 0-15 -> K {0..7,16..23}; lanes 16-31 -> K {8..15,24..31}
  const int lrow = lane & 15;
  const int kb   = (lane >> 4) * 8;

  // staging: A = one 32-elem row per thread (4x b128); B = 8 elems per thread
  const int brow = tid >> 2, bseg = (tid & 3) * 8;
  const unsigned short* gA = Xb + (size_t)(pBase + tid) * KQDIM;
  const unsigned short* gB = Wt + (size_t)(jBase + brow) * KQDIM + bseg;

  v8f acc[2][4] = {};

  auto stage = [&](int kq, int b) {
#pragma unroll
    for (int s = 0; s < 4; ++s)
      async_copy_b128(&As[b][tid * PITCH + s * 8], gA + kq + s * 8);
    async_copy_b128(&Bs[b][brow * PITCH + bseg], gB + kq);
  };

  stage(0, 0);
  wait_async0();
  __syncthreads();

  for (int s = 0; s < KQDIM / 32; ++s) {
    const int cur = s & 1;
    if (s + 1 < KQDIM / 32) stage((s + 1) * 32, cur ^ 1);  // prefetch next tile via async DMA

    FragBF af[2], bf[4];
#pragma unroll
    for (int i = 0; i < 2; ++i) {
      const unsigned short* base = &As[cur][(wm + i * 16 + lrow) * PITCH + kb];
      af[i].u[0] = *reinterpret_cast<const uint4*>(base);
      af[i].u[1] = *reinterpret_cast<const uint4*>(base + 16);
    }
#pragma unroll
    for (int j = 0; j < 4; ++j) {
      const unsigned short* base = &Bs[cur][(j * 16 + lrow) * PITCH + kb];
      bf[j].u[0] = *reinterpret_cast<const uint4*>(base);
      bf[j].u[1] = *reinterpret_cast<const uint4*>(base + 16);
    }
#pragma unroll
    for (int i = 0; i < 2; ++i)
#pragma unroll
      for (int j = 0; j < 4; ++j)
        acc[i][j] = __builtin_amdgcn_wmma_f32_16x16x32_bf16(
            false, af[i].v, false, bf[j].v, (short)0, acc[i][j], false, false);

    wait_async0();     // next tile's DMA done (overlapped with the 8 WMMAs above)
    __syncthreads();   // all waves done reading 'cur', all async writes visible
  }

  // epilogue: C-tile layout -> lane l: col = l&15, rows M = r + 8*(l>>4)
  const int rOff = (lane >> 4) * 8;
#pragma unroll
  for (int i = 0; i < 2; ++i) {
#pragma unroll
    for (int j = 0; j < 4; ++j) {
      int col = jBase + j * 16 + lrow;
      float bv = bias[col];
      int c = col / VJ, w = col - c * VJ;
#pragma unroll
      for (int r = 0; r < 8; ++r) {
        int p = pBase + wm + i * 16 + rOff + r;
        int n = p >> 8, t = p & 255;
        z[((size_t)(n * COUT + c) * TLEN + t) * VJ + w] = acc[i][j][r] + bv;
      }
    }
  }
}

// --- K3: deterministic per-channel sum / sumsq (one block per channel) -----
__global__ __launch_bounds__(512) void gcn_bn_stats(const float* __restrict__ z,
                                                    float* __restrict__ stats) {
  const int c   = blockIdx.x;
  const int tid = threadIdx.x;
  float s = 0.f, ss = 0.f;
  for (int i = tid; i < CHCNT; i += 512) {
    int n   = i / (TLEN * VJ);
    int rem = i - n * (TLEN * VJ);
    float v = z[(size_t)(n * COUT + c) * (TLEN * VJ) + rem];
    s += v; ss += v * v;
  }
  __shared__ float rs[512], rss[512];
  rs[tid] = s; rss[tid] = ss;
  __syncthreads();
  for (int off = 256; off > 0; off >>= 1) {
    if (tid < off) { rs[tid] += rs[tid + off]; rss[tid] += rss[tid + off]; }
    __syncthreads();
  }
  if (tid == 0) { stats[2 * c] = rs[0]; stats[2 * c + 1] = rss[0]; }
}

// --- K4: in-place BN (biased var) + affine + ReLU --------------------------
__global__ __launch_bounds__(256) void gcn_bn_apply(float* __restrict__ z,
                                                    const float* __restrict__ stats,
                                                    const float* __restrict__ gamma,
                                                    const float* __restrict__ beta) {
  size_t idx = (size_t)blockIdx.x * 256 + threadIdx.x;
  if (idx >= ELEMS) return;
  int c = (int)((idx / (TLEN * VJ)) & (COUT - 1));
  const float inv_cnt = 1.f / (float)CHCNT;
  float mean = stats[2 * c] * inv_cnt;
  float var  = stats[2 * c + 1] * inv_cnt - mean * mean;
  float sc   = rsqrtf(var + 1e-5f) * gamma[c];
  float val  = (z[idx] - mean) * sc + beta[c];
  z[idx] = val > 0.f ? val : 0.f;
}

// ---------------------------------------------------------------------------
extern "C" void kernel_launch(void* const* d_in, const int* in_sizes, int n_in,
                              void* d_out, int out_size, void* d_ws, size_t ws_size,
                              hipStream_t stream) {
  const float* x      = (const float*)d_in[0];
  const float* conv_w = (const float*)d_in[1];
  const float* conv_b = (const float*)d_in[2];
  const float* A      = (const float*)d_in[3];
  const float* gamma  = (const float*)d_in[4];
  const float* beta   = (const float*)d_in[5];
  (void)in_sizes; (void)n_in; (void)out_size; (void)ws_size;

  char* ws = (char*)d_ws;
  unsigned short* Xb = (unsigned short*)(ws);              // 16384*1600*2 = 52,428,800 B
  unsigned short* Wt = (unsigned short*)(ws + 52428800);   //  1600*1600*2 =  5,120,000 B
  float* bias        = (float*)(ws + 57548800);            //        1600*4 =     6,400 B
  float* stats       = (float*)(ws + 57555200);            //         128*4 =       512 B
  float* z           = (float*)d_out;                      // pre-BN z, then final output

  gcn_convert_x<<<(unsigned)((ELEMS + 255) / 256), 256, 0, stream>>>(x, Xb);
  gcn_build_weff<<<(NJDIM * KQDIM + 255) / 256, 256, 0, stream>>>(conv_w, conv_b, A, Wt, bias);
  dim3 gg(MDIM / BM, NJDIM / BN);   // (64, 25)
  gcn_gemm_wmma<<<gg, 256, 0, stream>>>(Xb, Wt, bias, z);
  gcn_bn_stats<<<COUT, 512, 0, stream>>>(z, stats);
  gcn_bn_apply<<<(unsigned)((ELEMS + 255) / 256), 256, 0, stream>>>(z, stats, gamma, beta);
}
